// Core_36567351558274
// MI455X (gfx1250) — compile-verified, matrix operand order checked by reading, not run on hardware
//
#include <hip/hip_runtime.h>
#include <hip/hip_bf16.h>
#include <math.h>

// Problem dims (from reference): T=8192, O=1024, H=2048, A=512
#define TT 8192
#define OO 1024
#define HH 2048
#define G3 6144   // 3*H
#define AO 512

typedef __bf16 bf16_t;
typedef __attribute__((ext_vector_type(16))) __bf16 v16bf;
typedef __attribute__((ext_vector_type(8)))  __bf16 v8bf;
typedef __attribute__((ext_vector_type(8)))  float  v8f;

// ---------------------------------------------------------------------------
// f32 -> bf16 conversion
// ---------------------------------------------------------------------------
__global__ void __launch_bounds__(256) cvt_f32_bf16(const float* __restrict__ x,
                                                    bf16_t* __restrict__ y,
                                                    int n) {
  int i = blockIdx.x * 256 + threadIdx.x;
  if (i < n) y[i] = (bf16_t)x[i];
}

// ---------------------------------------------------------------------------
// WMMA bf16 fragment load: 16 bf16 per lane, CDNA5 16-bit A/B layout.
// lane<16 : row m=lane,    K = {kb..kb+7} and {kb+16..kb+23}, kb = kk+0*8
// lane>=16: row m=lane-16, same pattern with kb = kk+8
// caller passes p already offset by (kk + half*8); grab 8 + 8 elements.
// ---------------------------------------------------------------------------
__device__ __forceinline__ v16bf load_frag16(const bf16_t* p) {
  union { uint4 u[2]; v16bf v; } t;
  t.u[0] = *(const uint4*)(p);       // K offsets +0..+7   (16 bytes)
  t.u[1] = *(const uint4*)(p + 16);  // K offsets +16..+23 (16 bytes)
  return t.v;
}

struct Frags { v16bf a, b0, b1, b2, b3; };

// ---------------------------------------------------------------------------
// gx = obs @ W_ih^T + b_ih    (TT x G3) = (TT x OO) * (G3 x OO)^T
// Both operands K-contiguous => natural NT WMMA. Each wave: 16(M) x 64(N),
// A fragment reused across 4 B fragments. Inner K loop is software-pipelined
// two-deep so v_wmma overlaps the next fragment set's global_load_b128s.
// ---------------------------------------------------------------------------
__global__ void __launch_bounds__(256) gemm_gx_bf16(
    const bf16_t* __restrict__ A,    // TT x OO
    const bf16_t* __restrict__ B,    // G3 x OO
    const float*  __restrict__ bias, // G3
    float* __restrict__ C)           // TT x G3
{
  const int K = OO, N = G3;
  int wave = blockIdx.x * (blockDim.x >> 5) + (threadIdx.x >> 5);
  int lane = threadIdx.x & 31;
  int half = lane >> 4;
  int m    = lane & 15;
  const int nJob = N / 64;               // 96 N-jobs
  int mi = wave / nJob;                  // 0..511
  int n0 = (wave - mi * nJob) * 64;

  const bf16_t* arow = A + (size_t)(mi * 16 + m) * K + half * 8;
  const bf16_t* b0r  = B + (size_t)(n0 +  0 + m) * K + half * 8;
  const bf16_t* b1r  = B + (size_t)(n0 + 16 + m) * K + half * 8;
  const bf16_t* b2r  = B + (size_t)(n0 + 32 + m) * K + half * 8;
  const bf16_t* b3r  = B + (size_t)(n0 + 48 + m) * K + half * 8;

  v8f acc0 = {}, acc1 = {}, acc2 = {}, acc3 = {};
  Frags cur, nxt;

  // prologue: fragments for kk = 0
  cur.a  = load_frag16(arow);
  cur.b0 = load_frag16(b0r);
  cur.b1 = load_frag16(b1r);
  cur.b2 = load_frag16(b2r);
  cur.b3 = load_frag16(b3r);

  #pragma unroll 1
  for (int kk = 0; kk < K - 64; kk += 64) {
    // issue loads for kk+32, then compute on cur
    nxt.a  = load_frag16(arow + kk + 32);
    nxt.b0 = load_frag16(b0r  + kk + 32);
    nxt.b1 = load_frag16(b1r  + kk + 32);
    nxt.b2 = load_frag16(b2r  + kk + 32);
    nxt.b3 = load_frag16(b3r  + kk + 32);
    acc0 = __builtin_amdgcn_wmma_f32_16x16x32_bf16(false, cur.a, false, cur.b0, (short)0, acc0, false, false);
    acc1 = __builtin_amdgcn_wmma_f32_16x16x32_bf16(false, cur.a, false, cur.b1, (short)0, acc1, false, false);
    acc2 = __builtin_amdgcn_wmma_f32_16x16x32_bf16(false, cur.a, false, cur.b2, (short)0, acc2, false, false);
    acc3 = __builtin_amdgcn_wmma_f32_16x16x32_bf16(false, cur.a, false, cur.b3, (short)0, acc3, false, false);
    // issue loads for kk+64, then compute on nxt
    cur.a  = load_frag16(arow + kk + 64);
    cur.b0 = load_frag16(b0r  + kk + 64);
    cur.b1 = load_frag16(b1r  + kk + 64);
    cur.b2 = load_frag16(b2r  + kk + 64);
    cur.b3 = load_frag16(b3r  + kk + 64);
    acc0 = __builtin_amdgcn_wmma_f32_16x16x32_bf16(false, nxt.a, false, nxt.b0, (short)0, acc0, false, false);
    acc1 = __builtin_amdgcn_wmma_f32_16x16x32_bf16(false, nxt.a, false, nxt.b1, (short)0, acc1, false, false);
    acc2 = __builtin_amdgcn_wmma_f32_16x16x32_bf16(false, nxt.a, false, nxt.b2, (short)0, acc2, false, false);
    acc3 = __builtin_amdgcn_wmma_f32_16x16x32_bf16(false, nxt.a, false, nxt.b3, (short)0, acc3, false, false);
  }
  // epilogue: kk = K-64 and K-32
  {
    nxt.a  = load_frag16(arow + K - 32);
    nxt.b0 = load_frag16(b0r  + K - 32);
    nxt.b1 = load_frag16(b1r  + K - 32);
    nxt.b2 = load_frag16(b2r  + K - 32);
    nxt.b3 = load_frag16(b3r  + K - 32);
    acc0 = __builtin_amdgcn_wmma_f32_16x16x32_bf16(false, cur.a, false, cur.b0, (short)0, acc0, false, false);
    acc1 = __builtin_amdgcn_wmma_f32_16x16x32_bf16(false, cur.a, false, cur.b1, (short)0, acc1, false, false);
    acc2 = __builtin_amdgcn_wmma_f32_16x16x32_bf16(false, cur.a, false, cur.b2, (short)0, acc2, false, false);
    acc3 = __builtin_amdgcn_wmma_f32_16x16x32_bf16(false, cur.a, false, cur.b3, (short)0, acc3, false, false);
    acc0 = __builtin_amdgcn_wmma_f32_16x16x32_bf16(false, nxt.a, false, nxt.b0, (short)0, acc0, false, false);
    acc1 = __builtin_amdgcn_wmma_f32_16x16x32_bf16(false, nxt.a, false, nxt.b1, (short)0, acc1, false, false);
    acc2 = __builtin_amdgcn_wmma_f32_16x16x32_bf16(false, nxt.a, false, nxt.b2, (short)0, acc2, false, false);
    acc3 = __builtin_amdgcn_wmma_f32_16x16x32_bf16(false, nxt.a, false, nxt.b3, (short)0, acc3, false, false);
  }

  // C/D layout: VGPR r -> row (r + half*8), col = n-base + (lane & 15)
  int rbase = mi * 16 + half * 8;
  {
    int col = n0 + m; float bb = bias[col];
    #pragma unroll
    for (int r = 0; r < 8; ++r) C[(size_t)(rbase + r) * N + col] = acc0[r] + bb;
  }
  {
    int col = n0 + 16 + m; float bb = bias[col];
    #pragma unroll
    for (int r = 0; r < 8; ++r) C[(size_t)(rbase + r) * N + col] = acc1[r] + bb;
  }
  {
    int col = n0 + 32 + m; float bb = bias[col];
    #pragma unroll
    for (int r = 0; r < 8; ++r) C[(size_t)(rbase + r) * N + col] = acc2[r] + bb;
  }
  {
    int col = n0 + 48 + m; float bb = bias[col];
    #pragma unroll
    for (int r = 0; r < 8; ++r) C[(size_t)(rbase + r) * N + col] = acc3[r] + bb;
  }
}

// ---------------------------------------------------------------------------
// Device-wide sense-reversing barrier for the persistent scan kernel.
// State (cnt at [0], gen at [16]) is zeroed by hipMemsetAsync each call.
// ---------------------------------------------------------------------------
__device__ __forceinline__ void grid_barrier(unsigned* cnt, volatile unsigned* gen) {
  __threadfence();          // make this thread's stores device-visible
  __syncthreads();
  if (threadIdx.x == 0) {
    unsigned g = *gen;
    unsigned arrived = atomicAdd(cnt, 1u);
    if (arrived == (unsigned)(gridDim.x - 1)) {
      atomicExch(cnt, 0u);
      __threadfence();
      atomicAdd((unsigned*)gen, 1u);
    } else {
      while (*gen == g) { __builtin_amdgcn_s_sleep(2); }
    }
    __threadfence();
  }
  __syncthreads();
}

// ---------------------------------------------------------------------------
// Persistent GRU scan. 128 blocks x 256 threads. Block b owns j = b*16..+15.
// LDS: 48 rows of W_hh (r/z/n gate rows for the 16 owned j) as bf16 = 192 KB,
// resident for all 8192 steps -> W_hh read from HBM exactly once.
// Per step: broadcast h_{t-1} (8 KB) into LDS, 16 lanes per j do partial dots
// with ds_load_b128-vectorized reads, shuffle-reduce, gate math, write hs[t],
// device barrier.
// ---------------------------------------------------------------------------
__global__ void __launch_bounds__(256) gru_scan(
    const float* __restrict__ gx,   // TT x G3  (includes b_ih)
    const float* __restrict__ Whh,  // G3 x HH
    const float* __restrict__ bhh,  // G3
    float* __restrict__ hs,         // TT x HH
    unsigned* __restrict__ bar)     // [0]=cnt, [16]=gen
{
  extern __shared__ unsigned char smem[];
  bf16_t* w   = (bf16_t*)smem;                      // 48 * 2048 bf16 = 192 KB
  float*  hsh = (float*)(smem + 48 * HH * 2);       // 2048 f32 = 8 KB

  const int tid = threadIdx.x;
  const int j0  = blockIdx.x * 16;

  // One-time: stage this block's 48 W_hh rows into LDS as bf16.
  for (int idx = tid; idx < 48 * HH; idx += 256) {
    int rrow = idx >> 11;          // 0..47  (gate g = rrow>>4, local j = rrow&15)
    int k    = idx & (HH - 1);
    int g = rrow >> 4, jj = rrow & 15;
    w[idx] = (bf16_t)Whh[(size_t)(g * HH + j0 + jj) * HH + k];
  }
  __syncthreads();

  const int jl   = tid >> 4;       // local j: 0..15
  const int part = tid & 15;       // k-partition: 0..15
  const int j    = j0 + jl;
  const float br = bhh[j], bz = bhh[HH + j], bn = bhh[2 * HH + j];
  const int kbase = part * 128;
  const v8bf* wrv = (const v8bf*)(w + (size_t)(0 * 16 + jl) * HH + kbase);
  const v8bf* wzv = (const v8bf*)(w + (size_t)(1 * 16 + jl) * HH + kbase);
  const v8bf* wnv = (const v8bf*)(w + (size_t)(2 * 16 + jl) * HH + kbase);
  const float4* hv4 = (const float4*)(hsh + kbase);

  unsigned* cnt = bar;
  volatile unsigned* gen = (volatile unsigned*)(bar + 16);

  for (int t = 0; t < TT; ++t) {
    // Stage h_{t-1} into LDS (zeros for t==0).
    if (t == 0) {
      for (int k = tid; k < HH; k += 256) hsh[k] = 0.f;
    } else {
      const float* hp = hs + (size_t)(t - 1) * HH;
      for (int k = tid; k < HH; k += 256) hsh[k] = hp[k];
    }
    __syncthreads();

    float sr = 0.f, sz = 0.f, sn = 0.f;
    #pragma unroll 4
    for (int i = 0; i < 16; ++i) {         // 16 chunks of 8 k-values
      v8bf wrx = wrv[i];
      v8bf wzx = wzv[i];
      v8bf wnx = wnv[i];
      float4 h0 = hv4[2 * i];
      float4 h1 = hv4[2 * i + 1];
      float hk[8] = { h0.x, h0.y, h0.z, h0.w, h1.x, h1.y, h1.z, h1.w };
      #pragma unroll
      for (int e = 0; e < 8; ++e) {
        float h = hk[e];
        sr += h * (float)wrx[e];
        sz += h * (float)wzx[e];
        sn += h * (float)wnx[e];
      }
    }
    // reduce the 16 k-partitions (xor<16 stays inside each 16-lane group)
    #pragma unroll
    for (int off = 8; off; off >>= 1) {
      sr += __shfl_xor(sr, off, 32);
      sz += __shfl_xor(sz, off, 32);
      sn += __shfl_xor(sn, off, 32);
    }

    if (part == 0) {
      const float* gxt = gx + (size_t)t * G3;
      float r   = 1.f / (1.f + __expf(-(gxt[j]          + br + sr)));
      float z   = 1.f / (1.f + __expf(-(gxt[HH + j]     + bz + sz)));
      float ghn = sn + bn;
      float n   = tanhf(gxt[2 * HH + j] + r * ghn);
      float hprev = hsh[j];
      hs[(size_t)t * HH + j] = (1.f - z) * n + z * hprev;
    }
    grid_barrier(cnt, gen);
  }
}

// ---------------------------------------------------------------------------
// Attention: scores[t] = hs[t] . hs[T-1], t in [0, T-1)
// ---------------------------------------------------------------------------
__global__ void __launch_bounds__(256) att_scores(const float* __restrict__ hs,
                                                  float* __restrict__ scores) {
  __shared__ float red[256];
  const float* a = hs + (size_t)blockIdx.x * HH;
  const float* l = hs + (size_t)(TT - 1) * HH;
  float s = 0.f;
  for (int k = threadIdx.x; k < HH; k += 256) s += a[k] * l[k];
  red[threadIdx.x] = s; __syncthreads();
  for (int o = 128; o; o >>= 1) {
    if (threadIdx.x < o) red[threadIdx.x] += red[threadIdx.x + o];
    __syncthreads();
  }
  if (threadIdx.x == 0) scores[blockIdx.x] = red[0];
}

// softmax over 8191 scores (in place -> exp weights); stats[0] = 1/sum
__global__ void __launch_bounds__(1024) att_softmax(float* __restrict__ scores,
                                                    float* __restrict__ stats) {
  const int N = TT - 1;
  __shared__ float red[1024];
  int tid = threadIdx.x;
  float mx = -INFINITY;
  for (int i = tid; i < N; i += 1024) mx = fmaxf(mx, scores[i]);
  red[tid] = mx; __syncthreads();
  for (int o = 512; o; o >>= 1) {
    if (tid < o) red[tid] = fmaxf(red[tid], red[tid + o]);
    __syncthreads();
  }
  mx = red[0]; __syncthreads();
  float sum = 0.f;
  for (int i = tid; i < N; i += 1024) {
    float e = __expf(scores[i] - mx);
    scores[i] = e;
    sum += e;
  }
  red[tid] = sum; __syncthreads();
  for (int o = 512; o; o >>= 1) {
    if (tid < o) red[tid] += red[tid + o];
    __syncthreads();
  }
  if (tid == 0) stats[0] = 1.f / red[0];
}

// c[h] = (1/sum) * sum_t attn[t] * hs[t][h]   (column sweep, coalesced in h)
__global__ void __launch_bounds__(256) att_context(const float* __restrict__ hs,
                                                   const float* __restrict__ attn,
                                                   const float* __restrict__ stats,
                                                   float* __restrict__ c) {
  const int N = TT - 1;
  int h = blockIdx.x * 256 + threadIdx.x;
  float acc = 0.f;
  for (int t = 0; t < N; ++t) acc += attn[t] * hs[(size_t)t * HH + h];
  c[h] = acc * stats[0];
}

// out[0..511] = c @ W_o^T + b_o ; out[512..1023] = c @ W_d^T + b_d
__global__ void __launch_bounds__(256) out_logits(const float* __restrict__ c,
                                                  const float* __restrict__ Wo,
                                                  const float* __restrict__ bo,
                                                  const float* __restrict__ Wd,
                                                  const float* __restrict__ bd,
                                                  float* __restrict__ out) {
  __shared__ float red[256];
  int row = blockIdx.x;                // 0..1023
  const float* Wrow; float bb;
  if (row < AO) { Wrow = Wo + (size_t)row * HH;        bb = bo[row]; }
  else          { Wrow = Wd + (size_t)(row - AO) * HH; bb = bd[row - AO]; }
  float s = 0.f;
  for (int k = threadIdx.x; k < HH; k += 256) s += c[k] * Wrow[k];
  red[threadIdx.x] = s; __syncthreads();
  for (int o = 128; o; o >>= 1) {
    if (threadIdx.x < o) red[threadIdx.x] += red[threadIdx.x + o];
    __syncthreads();
  }
  if (threadIdx.x == 0) out[row] = red[0] + bb;
}

// ---------------------------------------------------------------------------
// Host launcher
// ---------------------------------------------------------------------------
extern "C" void kernel_launch(void* const* d_in, const int* in_sizes, int n_in,
                              void* d_out, int out_size, void* d_ws, size_t ws_size,
                              hipStream_t stream) {
  const float* obs  = (const float*)d_in[0];   // TT x OO
  const float* W_ih = (const float*)d_in[1];   // G3 x OO
  const float* W_hh = (const float*)d_in[2];   // G3 x HH
  const float* b_ih = (const float*)d_in[3];   // G3
  const float* b_hh = (const float*)d_in[4];   // G3
  const float* W_o  = (const float*)d_in[5];   // AO x HH
  const float* b_o  = (const float*)d_in[6];   // AO
  const float* W_d  = (const float*)d_in[7];   // AO x HH
  const float* b_d  = (const float*)d_in[8];   // AO
  float* out = (float*)d_out;

  char* ws = (char*)d_ws;
  // 256-aligned workspace carve-up
  float*  gx     = (float*)(ws);                                      // 201326592 B
  float*  hs     = (float*)(ws + 201326592ull);                       //  67108864 B
  bf16_t* obs_bf = (bf16_t*)(ws + 268435456ull);                      //  16777216 B
  bf16_t* wih_bf = (bf16_t*)(ws + 285212672ull);                      //  12582912 B
  float*  scores = (float*)(ws + 297795584ull);                       //     32768 B
  float*  cvec   = (float*)(ws + 297828352ull);                       //      8192 B
  float*  stats  = (float*)(ws + 297836544ull);                       //       256 B
  unsigned* bar  = (unsigned*)(ws + 297836800ull);                    //       256 B

  // 1) bf16 copies of obs and W_ih for WMMA
  {
    int n = TT * OO;
    cvt_f32_bf16<<<(n + 255) / 256, 256, 0, stream>>>(obs, obs_bf, n);
  }
  {
    int n = G3 * OO;
    cvt_f32_bf16<<<(n + 255) / 256, 256, 0, stream>>>(W_ih, wih_bf, n);
  }

  // 2) gx = obs @ W_ih^T + b_ih via WMMA (512 M-tiles x 96 N-jobs, 8 waves/blk)
  {
    int nWaves = (TT / 16) * (G3 / 64);   // 49152
    gemm_gx_bf16<<<nWaves / 8, 256, 0, stream>>>(obs_bf, wih_bf, b_ih, gx);
  }

  // 3) sequential GRU scan (persistent kernel, W_hh pinned in LDS)
  hipMemsetAsync(bar, 0, 256, stream);
  {
    const size_t smem = (size_t)48 * HH * 2 + (size_t)HH * 4;  // 204800 B
    hipFuncSetAttribute((const void*)gru_scan,
                        hipFuncAttributeMaxDynamicSharedMemorySize, (int)smem);
    gru_scan<<<HH / 16, 256, smem, stream>>>(gx, W_hh, b_hh, hs, bar);
  }

  // 4) attention pool
  att_scores<<<TT - 1, 256, 0, stream>>>(hs, scores);
  att_softmax<<<1, 1024, 0, stream>>>(scores, stats);
  att_context<<<HH / 256, 256, 0, stream>>>(hs, scores, stats, cvec);

  // 5) output heads
  out_logits<<<2 * AO, 256, 0, stream>>>(cvec, W_o, b_o, W_d, b_d, out);
}
